// MultiScaleEvidentialDecoder_67723044323884
// MI455X (gfx1250) — matrix operand order repaired and checked by reading, not compiled, not run on hardware
//
#include <hip/hip_runtime.h>
#include <cstdint>

typedef __attribute__((ext_vector_type(16))) _Float16 v16h;
typedef __attribute__((ext_vector_type(8)))  float    v8f;

#define NB   16      // batch
#define HF   128     // feature spatial
#define NPIX (HF*HF) // 16384
#define HB   32
#define LTOK (HB*HB) // 1024
#define TOPK 12
#define PSTR 80      // LDS pixel stride (64 ch + 16 pad), 160B = 32B-aligned

__device__ __forceinline__ float softplus_f(float x) {
    return (x > 20.f) ? x : log1pf(__expf(x));
}

// A-fragment block order: position j in a 32-K block reads source k = j
//   (contiguous, matching the B fragments)
__device__ __forceinline__ int permA(int j) {
    return j;
}

// ---------------------------------------------------------------------------
// Fold: x [B, L=1024, 1024] -> x_feat f32/f16 [B, 64, 128, 128]
// ---------------------------------------------------------------------------
__global__ void __launch_bounds__(256) fold_kernel(const float* __restrict__ x,
                                                   float* __restrict__ xf,
                                                   _Float16* __restrict__ xh)
{
    int i = blockIdx.x * 256 + threadIdx.x;        // over 16*64*16384
    int p  = i & (NPIX - 1);
    int bc = i >> 14;
    int c  = bc & 63;
    int b  = bc >> 6;
    int y  = p >> 7, xp = p & 127;
    int hb = y >> 2, r = y & 3, wb = xp >> 2, s = xp & 3;
    int l  = hb * HB + wb;
    int d  = c * 16 + r * 4 + s;
    float v = x[((size_t)b * LTOK + l) * 1024 + d];
    xf[i] = v;
    xh[i] = (_Float16)v;
}

// ---------------------------------------------------------------------------
// Bilinear resize (half-pixel, edge clamp), f32 in -> f16 out plane(s)
// ---------------------------------------------------------------------------
__global__ void __launch_bounds__(256) bilinear_kernel(const float* __restrict__ in,
                                                       int C, int Hin, int Win,
                                                       int Hout, int Wout,
                                                       _Float16* __restrict__ out,
                                                       int outChTot, int chOff)
{
    int p = blockIdx.x * 256 + threadIdx.x;
    int np = Hout * Wout;
    if (p >= np) return;
    int plane = blockIdx.y;
    int b = plane / C, c = plane - b * C;
    const float* ip = in + (size_t)plane * Hin * Win;
    int oy = p / Wout, ox = p - oy * Wout;
    float fy = (oy + 0.5f) * (float)Hin / (float)Hout - 0.5f;
    float fx = (ox + 0.5f) * (float)Win / (float)Wout - 0.5f;
    fy = fminf(fmaxf(fy, 0.f), (float)(Hin - 1));
    fx = fminf(fmaxf(fx, 0.f), (float)(Win - 1));
    int y0 = (int)fy, x0 = (int)fx;
    float wy = fy - y0, wx = fx - x0;
    int y1 = min(y0 + 1, Hin - 1), x1 = min(x0 + 1, Win - 1);
    float v00 = ip[y0 * Win + x0], v01 = ip[y0 * Win + x1];
    float v10 = ip[y1 * Win + x0], v11 = ip[y1 * Win + x1];
    float v = v00 * (1 - wy) * (1 - wx) + v01 * (1 - wy) * wx
            + v10 * wy * (1 - wx) + v11 * wy * wx;
    out[((size_t)(b * outChTot + chOff + c)) * np + p] = (_Float16)v;
}

// ---------------------------------------------------------------------------
// 3x3 conv 64->64 + bias + relu, implicit GEMM via v_wmma_f32_16x16x32_f16.
// K ordered as k = (dy*3+dx)*64 + c  -> each 32-K step has uniform (dy,dx).
// LDS tile is channel-innermost (sIn[pixPad*PSTR + c]) so a B fragment is one
// contiguous 32B LDS load; weights are block-permuted so an A fragment is one
// contiguous 32B global load.
// ---------------------------------------------------------------------------
__global__ void __launch_bounds__(256) conv3x3_wmma(const _Float16* __restrict__ in,
                                                    const _Float16* __restrict__ wk,
                                                    const float* __restrict__ bias,
                                                    _Float16* __restrict__ out,
                                                    int H, int W, int rows, int cols)
{
    __shared__ _Float16 sIn[198 * PSTR];   // max padded tile (rows=1,cols=64): 31680 B

    int b  = blockIdx.z;
    int y0 = blockIdx.y * rows;
    int x0 = blockIdx.x * cols;
    int tid = threadIdx.x;
    int colsP = cols + 2, area = (rows + 2) * colsP;

    const _Float16* inB = in + (size_t)b * 64 * H * W;
    int tot = 64 * area;
    for (int i = tid; i < tot; i += 256) {
        int c = i / area; int rem = i - c * area;        // rem = ry*colsP + cx
        int ry = rem / colsP; int cx = rem - ry * colsP;
        int yy = y0 + ry - 1, xx = x0 + cx - 1;
        _Float16 v = (_Float16)0.f;
        if (yy >= 0 && yy < H && xx >= 0 && xx < W)
            v = inB[(c * H + yy) * W + xx];
        sIn[rem * PSTR + c] = v;
    }
    __syncthreads();

    int wave = tid >> 5, lane = tid & 31;
    int tileRow = wave & 3;
    int colBase = (wave >> 2) * 2;
    int hi = lane >> 4, m = lane & 15, n = lane & 15;

    int q0 = colBase * 16 + n;
    int q1 = q0 + 16;
    int ly0 = q0 / cols, lx0 = q0 - ly0 * cols;
    int ly1 = q1 / cols, lx1 = q1 - ly1 * cols;

    const _Float16* pB0 = sIn + (ly0 * colsP + lx0) * PSTR + hi * 16;
    const _Float16* pB1 = sIn + (ly1 * colsP + lx1) * PSTR + hi * 16;
    const _Float16* pA  = wk + (size_t)(tileRow * 16 + m) * 576 + hi * 16;

    v8f acc0 = {0.f,0.f,0.f,0.f,0.f,0.f,0.f,0.f};
    v8f acc1 = {0.f,0.f,0.f,0.f,0.f,0.f,0.f,0.f};

    #pragma unroll
    for (int kk = 0; kk < 18; ++kk) {
        int g  = kk >> 1;              // tap index 0..8 (compile-time after unroll)
        int dy = g / 3, dx = g - dy * 3;
        int bu = (dy * colsP + dx) * PSTR + (kk & 1) * 32;   // LDS offset
        v16h a   = *(const v16h*)(pA + kk * 32);
        v16h bf0 = *(const v16h*)(pB0 + bu);
        v16h bf1 = *(const v16h*)(pB1 + bu);
        acc0 = __builtin_amdgcn_wmma_f32_16x16x32_f16(false, a, false, bf0,
                                                      (short)0, acc0, false, false);
        acc1 = __builtin_amdgcn_wmma_f32_16x16x32_f16(false, a, false, bf1,
                                                      (short)0, acc1, false, false);
    }

    #pragma unroll
    for (int j = 0; j < 8; ++j) {
        int oc = tileRow * 16 + j + 8 * hi;
        float bv = bias[oc];
        {
            int yy = y0 + ly0, xx = x0 + lx0;
            float vv = fmaxf(acc0[j] + bv, 0.f);
            out[((size_t)(b * 64 + oc) * H + yy) * W + xx] = (_Float16)vv;
        }
        {
            int yy = y0 + ly1, xx = x0 + lx1;
            float vv = fmaxf(acc1[j] + bv, 0.f);
            out[((size_t)(b * 64 + oc) * H + yy) * W + xx] = (_Float16)vv;
        }
    }
}

// ---------------------------------------------------------------------------
// 1x1 conv GEMM via WMMA.  in: [B][Kc][Npix] f16, wk: [Cout][Kc] f16 (A-permuted).
// LDS is pixel-major / channel-innermost: B fragment = one 32B LDS load.
// ---------------------------------------------------------------------------
__global__ void __launch_bounds__(256) conv1x1_wmma(const _Float16* __restrict__ in,
                                                    const _Float16* __restrict__ wk,
                                                    const float* __restrict__ bias,
                                                    _Float16* __restrict__ outH,
                                                    float* __restrict__ outF,
                                                    int Kc, int Cout, int Npix,
                                                    int outBStride, int act)
{
    __shared__ _Float16 sIn[64 * 144];   // max: pixWG=64 x (Kc=64+16) or 32 x (128+16)

    int b = blockIdx.y;
    int TR = Cout >> 4;          // 2 or 4
    int TC = 8 / TR;             // 4 or 2
    int pixWG = TC * 16;         // 64 or 32
    int KcPad = Kc + 16;         // keep 32B row alignment, spread banks
    int p0 = blockIdx.x * pixWG;
    int tid = threadIdx.x;

    const _Float16* inB = in + (size_t)b * Kc * Npix;
    int tot = Kc * pixWG;
    for (int i = tid; i < tot; i += 256) {
        int c = i / pixWG; int p = i - c * pixWG;     // coalesced global read
        sIn[p * KcPad + c] = inB[(size_t)c * Npix + p0 + p];
    }
    __syncthreads();

    int wave = tid >> 5, lane = tid & 31;
    int tileRow = wave % TR;
    int tileCol = wave / TR;
    int hi = lane >> 4, m = lane & 15, n = lane & 15;

    const _Float16* pB = sIn + (tileCol * 16 + n) * KcPad + hi * 16;
    const _Float16* pA = wk + (size_t)(tileRow * 16 + m) * Kc + hi * 16;

    v8f acc = {0.f,0.f,0.f,0.f,0.f,0.f,0.f,0.f};

    int ksteps = Kc >> 5;
    #pragma unroll 4
    for (int kk = 0; kk < ksteps; ++kk) {
        v16h a  = *(const v16h*)(pA + kk * 32);
        v16h bf = *(const v16h*)(pB + kk * 32);
        acc = __builtin_amdgcn_wmma_f32_16x16x32_f16(false, a, false, bf,
                                                     (short)0, acc, false, false);
    }

    #pragma unroll
    for (int j = 0; j < 8; ++j) {
        int oc = tileRow * 16 + j + 8 * hi;
        float vv = acc[j] + bias[oc];
        vv = (act == 1) ? fmaxf(vv, 0.f) : softplus_f(vv);
        int p = p0 + tileCol * 16 + n;
        if (outH) outH[(size_t)b * outBStride + (size_t)oc * Npix + p] = (_Float16)vv;
        else      outF[(size_t)b * outBStride + (size_t)oc * Npix + p] = vv;
    }
}

// ---------------------------------------------------------------------------
// channel mean over [B*64] planes of 16384
// ---------------------------------------------------------------------------
__global__ void __launch_bounds__(256) channel_mean_kernel(const float* __restrict__ xf,
                                                           float* __restrict__ means)
{
    __shared__ float sred[256];
    int bc = blockIdx.x;
    const float* p = xf + (size_t)bc * NPIX;
    float s = 0.f;
    for (int i = threadIdx.x; i < NPIX; i += 256) s += p[i];
    sred[threadIdx.x] = s;
    __syncthreads();
    for (int st = 128; st > 0; st >>= 1) {
        if (threadIdx.x < st) sred[threadIdx.x] += sred[threadIdx.x + st];
        __syncthreads();
    }
    if (threadIdx.x == 0) means[bc] = sred[0] * (1.f / NPIX);
}

__global__ void topk_kernel(const float* __restrict__ means, int* __restrict__ idxOut)
{
    int b = blockIdx.x;
    if (threadIdx.x != 0) return;
    float m[64];
    for (int c = 0; c < 64; ++c) m[c] = means[b * 64 + c];
    for (int j = 0; j < TOPK; ++j) {
        int am = 0; float mv = m[0];
        for (int c = 1; c < 64; ++c) if (m[c] > mv) { mv = m[c]; am = c; }
        idxOut[b * TOPK + j] = am;
        m[am] = -1e30f;
    }
}

__global__ void init_small(unsigned* __restrict__ mx, unsigned* __restrict__ minmax)
{
    int t = threadIdx.x;
    if (t < NB * TOPK) mx[t] = 0u;
    if (t == 0) { minmax[0] = 0x7f800000u; minmax[1] = 0u; }
}

// ---------------------------------------------------------------------------
// sobel magnitude on the 12 selected channels + per-(b,j) max
// ---------------------------------------------------------------------------
__global__ void __launch_bounds__(256) sobel_sel_kernel(const float* __restrict__ xf,
                                                        const int* __restrict__ idxArr,
                                                        float* __restrict__ sel,
                                                        unsigned* __restrict__ mx)
{
    __shared__ float sm[256];
    int p = blockIdx.x * 256 + threadIdx.x;
    int j = blockIdx.y, b = blockIdx.z;
    int c = idxArr[b * TOPK + j];
    const float* plane = xf + (size_t)(b * 64 + c) * NPIX;
    int y = p >> 7, xp = p & 127;
    const float kxw[3][3] = {{-1,0,1},{-2,0,2},{-1,0,1}};
    const float kyw[3][3] = {{-1,-2,-1},{0,0,0},{1,2,1}};
    float gx = 0.f, gy = 0.f;
    #pragma unroll
    for (int dy = 0; dy < 3; ++dy)
        #pragma unroll
        for (int dx = 0; dx < 3; ++dx) {
            int yy = y + dy - 1, xx = xp + dx - 1;
            if (yy >= 0 && yy < HF && xx >= 0 && xx < HF) {
                float v = plane[yy * HF + xx];
                gx += kxw[dy][dx] * v;
                gy += kyw[dy][dx] * v;
            }
        }
    float mag = sqrtf(gx * gx + gy * gy + 1e-8f);
    sel[(size_t)(b * TOPK + j) * NPIX + p] = mag;
    sm[threadIdx.x] = mag;
    __syncthreads();
    for (int st = 128; st > 0; st >>= 1) {
        if (threadIdx.x < st) sm[threadIdx.x] = fmaxf(sm[threadIdx.x], sm[threadIdx.x + st]);
        __syncthreads();
    }
    if (threadIdx.x == 0)
        atomicMax(&mx[b * TOPK + j], __float_as_uint(sm[0]));
}

// ---------------------------------------------------------------------------
// boundary map + global min/max
// ---------------------------------------------------------------------------
__global__ void __launch_bounds__(256) bm_kernel(const float* __restrict__ sel,
                                                 const unsigned* __restrict__ mx,
                                                 const float* __restrict__ scale_p,
                                                 float* __restrict__ bm,
                                                 unsigned* __restrict__ minmax)
{
    __shared__ float smin[256], smax[256];
    int p = blockIdx.x * 256 + threadIdx.x;
    int b = blockIdx.y;
    const float* sp = sel + (unsigned)(b * TOPK) * NPIX + p;
    float s = 0.f;
    #pragma unroll
    for (int j = 0; j < TOPK; ++j) {
        float mxv = __uint_as_float(mx[b * TOPK + j]) + 1e-8f;
        s += *sp / (mxv * 1.2f);
        sp += NPIX;
    }
    float bmv = s * (1.f / TOPK);
    float t = fminf(bmv * scale_p[0], 4.f);
    float v = 1.f / (1.f + __expf(-t));
    bm[(unsigned)b * NPIX + p] = v;
    smin[threadIdx.x] = v; smax[threadIdx.x] = v;
    __syncthreads();
    for (int st = 128; st > 0; st >>= 1) {
        if (threadIdx.x < st) {
            smin[threadIdx.x] = fminf(smin[threadIdx.x], smin[threadIdx.x + st]);
            smax[threadIdx.x] = fmaxf(smax[threadIdx.x], smax[threadIdx.x + st]);
        }
        __syncthreads();
    }
    if (threadIdx.x == 0) {
        atomicMin(&minmax[0], __float_as_uint(smin[0]));
        atomicMax(&minmax[1], __float_as_uint(smax[0]));
    }
}

__global__ void __launch_bounds__(256) bm_finalize_kernel(float* __restrict__ bm,
                                                          const unsigned* __restrict__ minmax)
{
    int i = blockIdx.x * 256 + threadIdx.x;     // over NB*NPIX
    float mn  = __uint_as_float(minmax[0]);
    float mxg = __uint_as_float(minmax[1]);
    float dr = mxg - mn;
    float blend = fminf(0.8f, 0.3f + (0.3f - dr) * 2.f);
    float v = bm[i];
    if (dr < 0.3f) {
        float st = (v - mn) / (dr + 1e-8f);
        v = (1.f - blend) * v + blend * st;
    }
    bm[i] = powf(v * 0.82f, 0.88f);
}

// ---------------------------------------------------------------------------
// bw3: 32->2 1x1 + softplus, * (1+1.5*bm), write f16 into fcat ch 96/97
// ---------------------------------------------------------------------------
__global__ void __launch_bounds__(256) bev3_kernel(const _Float16* __restrict__ bevr2,
                                                   const float* __restrict__ bw3,
                                                   const float* __restrict__ bb3,
                                                   const float* __restrict__ bm,
                                                   _Float16* __restrict__ fcat)
{
    int p = blockIdx.x * 256 + threadIdx.x;
    int b = blockIdx.y;
    const _Float16* ip = bevr2 + (unsigned)(b * 32) * NPIX + p;
    float a0 = bb3[0], a1 = bb3[1];
    #pragma unroll
    for (int c = 0; c < 32; ++c) {
        float v = (float)*ip;  ip += NPIX;
        a0 += bw3[c] * v;
        a1 += bw3[32 + c] * v;
    }
    a0 = softplus_f(a0);
    a1 = softplus_f(a1);
    float g = 1.f + 1.5f * bm[(unsigned)b * NPIX + p];
    _Float16* op = fcat + ((unsigned)b * 128 + 96) * NPIX + p;
    op[0]    = (_Float16)(a0 * g);
    op[NPIX] = (_Float16)(a1 * g);
}

// zero fcat channels 98..127 (padding for the K=128 fusion GEMM)
__global__ void __launch_bounds__(256) zero_fcat_pad(_Float16* __restrict__ fcat)
{
    int i = blockIdx.x * 256 + threadIdx.x;     // 16*30*16384
    int p = i & (NPIX - 1);
    int t = i >> 14;
    int ch = 98 + t % 30;
    int b  = t / 30;
    fcat[((unsigned)b * 128 + ch) * NPIX + p] = (_Float16)0.f;
}

// ---------------------------------------------------------------------------
// final: fw2 64->2 softplus, Dirichlet, prob out, weighted unfold -> tok
// ---------------------------------------------------------------------------
__global__ void __launch_bounds__(256) final_kernel(const _Float16* __restrict__ ffeat,
                                                    const float* __restrict__ fw2,
                                                    const float* __restrict__ fb2,
                                                    const float* __restrict__ uwp,
                                                    const float* __restrict__ xf,
                                                    float* __restrict__ outProb,
                                                    float* __restrict__ outTok)
{
    int p = blockIdx.x * 256 + threadIdx.x;
    int b = blockIdx.y;
    const _Float16* ip = ffeat + (unsigned)(b * 64) * NPIX + p;
    float e0 = fb2[0], e1 = fb2[1];
    #pragma unroll
    for (int c = 0; c < 64; ++c) {
        float v = (float)*ip;  ip += NPIX;
        e0 += fw2[c] * v;
        e1 += fw2[64 + c] * v;
    }
    e0 = softplus_f(e0);
    e1 = softplus_f(e1);
    float a0 = e0 + 1.f, a1 = e1 + 1.f;
    float S = a0 + a1;
    outProb[(unsigned)b * NPIX + p] = a1 / S;
    float alea = 2.f / S;
    float d = a0 - a1;
    float epi = 0.5f * d * d / (S * S);
    float unc = uwp[0] * alea + uwp[1] * epi;
    float w = 1.f - unc;
    int y = p >> 7, xp = p & 127;
    int hb = y >> 2, r = y & 3, wb = xp >> 2, s = xp & 3;
    float* tp = outTok + ((unsigned)(b * LTOK + hb * HB + wb)) * 1024 + r * 4 + s;
    const float* xb = xf + (unsigned)(b * 64) * NPIX + p;
    #pragma unroll
    for (int c = 0; c < 64; ++c) {
        *tp = w * *xb;
        tp += 16;  xb += NPIX;
    }
}

// ---------------------------------------------------------------------------
// weight casts (with A-fragment block permutation + im2col k-reorder)
// ---------------------------------------------------------------------------
__global__ void __launch_bounds__(256) cast_w3x3_perm(const float* __restrict__ in,
                                                      _Float16* __restrict__ out, int nO)
{
    int i = blockIdx.x * 256 + threadIdx.x;      // over nO*576
    if (i >= nO * 576) return;
    int o = i / 576, pos = i - o * 576;
    int kk = pos >> 5, j = pos & 31;
    int k = kk * 32 + permA(j);                  // logical k = (dy*3+dx)*64 + c
    int g = k >> 6, c = k & 63;
    int dy = g / 3, dx = g - dy * 3;
    out[i] = (_Float16)in[((o * 64 + c) * 3 + dy) * 3 + dx];
}

__global__ void __launch_bounds__(256) cast_w1x1_perm(const float* __restrict__ in,
                                                      _Float16* __restrict__ out,
                                                      int nO, int cin, int Kc)
{
    int i = blockIdx.x * 256 + threadIdx.x;      // over nO*Kc
    if (i >= nO * Kc) return;
    int o = i / Kc, pos = i - o * Kc;
    int kk = pos >> 5, j = pos & 31;
    int k = kk * 32 + permA(j);
    out[i] = (k < cin) ? (_Float16)in[o * cin + k] : (_Float16)0.f;
}

// ===========================================================================
extern "C" void kernel_launch(void* const* d_in, const int* in_sizes, int n_in,
                              void* d_out, int out_size, void* d_ws, size_t ws_size,
                              hipStream_t stream)
{
    (void)in_sizes; (void)n_in; (void)out_size; (void)ws_size;

    const float* x       = (const float*)d_in[0];
    const float* pyr_w1  = (const float*)d_in[1];
    const float* pyr_b1  = (const float*)d_in[2];
    const float* pyr_w2  = (const float*)d_in[3];
    const float* pyr_b2  = (const float*)d_in[4];
    const float* bw1     = (const float*)d_in[5];
    const float* bb1     = (const float*)d_in[6];
    const float* bw2     = (const float*)d_in[7];
    const float* bb2     = (const float*)d_in[8];
    const float* bw3     = (const float*)d_in[9];
    const float* bb3     = (const float*)d_in[10];
    const float* fw1     = (const float*)d_in[11];
    const float* fb1     = (const float*)d_in[12];
    const float* fw2     = (const float*)d_in[13];
    const float* fb2     = (const float*)d_in[14];
    const float* uw      = (const float*)d_in[15];
    const float* scale_p = (const float*)d_in[16];

    char* ws = (char*)d_ws;
    size_t off = 0;
    auto alloc = [&](size_t bytes) -> char* {
        off = (off + 255) & ~(size_t)255;
        char* p = ws + off;
        off += bytes;
        return p;
    };

    float*    xf32  = (float*)   alloc((size_t)NB*64*NPIX*4);       // 64 MB
    _Float16* xf16  = (_Float16*)alloc((size_t)NB*64*NPIX*2);       // 32 MB
    _Float16* fcat  = (_Float16*)alloc((size_t)NB*128*NPIX*2);      // 64 MB
    _Float16* r0h   = (_Float16*)alloc((size_t)NB*64*NPIX*2);       // 32 MB (reused)
    _Float16* x2h   = (_Float16*)alloc((size_t)NB*64*64*64*2);
    _Float16* x4h   = (_Float16*)alloc((size_t)NB*64*32*32*2);
    _Float16* r1h   = (_Float16*)alloc((size_t)NB*64*64*64*2);
    _Float16* r2h   = (_Float16*)alloc((size_t)NB*64*32*32*2);
    float*    e1f   = (float*)   alloc((size_t)NB*32*64*64*4);
    float*    e2f   = (float*)   alloc((size_t)NB*32*32*32*4);
    _Float16* bevr2 = (_Float16*)alloc((size_t)NB*32*NPIX*2);       // 16 MB
    float*    selb  = (float*)   alloc((size_t)NB*TOPK*NPIX*4);     // 12 MB
    float*    bmb   = (float*)   alloc((size_t)NB*NPIX*4);
    float*    means = (float*)   alloc((size_t)NB*64*4);
    int*      idxb  = (int*)     alloc((size_t)NB*TOPK*4);
    unsigned* mxb   = (unsigned*)alloc((size_t)NB*TOPK*4);
    unsigned* minmax= (unsigned*)alloc(2*4);
    _Float16* wkPyr = (_Float16*)alloc((size_t)3*64*576*2);
    _Float16* wkBw1 = (_Float16*)alloc((size_t)64*576*2);
    _Float16* wkPyr2= (_Float16*)alloc((size_t)3*32*64*2);
    _Float16* wkBw2 = (_Float16*)alloc((size_t)32*64*2);
    _Float16* wkFw1 = (_Float16*)alloc((size_t)64*128*2);

    float* outProb = (float*)d_out;
    float* outTok  = outProb + (size_t)NB * NPIX;

    // --- weight casts (permuted for WMMA A-fragment contiguous loads) ---
    cast_w3x3_perm<<<(3*64*576 + 255)/256, 256, 0, stream>>>(pyr_w1, wkPyr, 3*64);
    cast_w3x3_perm<<<(64*576   + 255)/256, 256, 0, stream>>>(bw1,    wkBw1, 64);
    cast_w1x1_perm<<<(3*32*64  + 255)/256, 256, 0, stream>>>(pyr_w2, wkPyr2, 3*32, 64, 64);
    cast_w1x1_perm<<<(32*64    + 255)/256, 256, 0, stream>>>(bw2,    wkBw2, 32, 64, 64);
    cast_w1x1_perm<<<(64*128   + 255)/256, 256, 0, stream>>>(fw1,    wkFw1, 64, 98, 128);

    zero_fcat_pad<<<(NB*30*NPIX)/256, 256, 0, stream>>>(fcat);
    init_small<<<1, 256, 0, stream>>>(mxb, minmax);

    // --- fold ---
    fold_kernel<<<(NB*64*NPIX)/256, 256, 0, stream>>>(x, xf32, xf16);

    // --- pyramid downsamples ---
    bilinear_kernel<<<dim3((64*64+255)/256, NB*64), 256, 0, stream>>>(
        xf32, 64, HF, HF, 64, 64, x2h, 64, 0);
    bilinear_kernel<<<dim3((32*32+255)/256, NB*64), 256, 0, stream>>>(
        xf32, 64, HF, HF, 32, 32, x4h, 64, 0);

    // --- pyramid 3x3 convs (relu) ---
    conv3x3_wmma<<<dim3(2, 128, NB), 256, 0, stream>>>(
        xf16, wkPyr + 0*64*576, pyr_b1 + 0*64, r0h, HF, HF, 1, 64);
    conv3x3_wmma<<<dim3(1, 64, NB), 256, 0, stream>>>(
        x2h,  wkPyr + 1*64*576, pyr_b1 + 1*64, r1h, 64, 64, 1, 64);
    conv3x3_wmma<<<dim3(1, 16, NB), 256, 0, stream>>>(
        x4h,  wkPyr + 2*64*576, pyr_b1 + 2*64, r2h, 32, 32, 2, 32);

    // --- pyramid 1x1 convs (softplus) ---
    conv1x1_wmma<<<dim3(NPIX/64, NB), 256, 0, stream>>>(
        r0h, wkPyr2 + 0*32*64, pyr_b2 + 0*32, fcat, nullptr,
        64, 32, NPIX, 128*NPIX, 2);                           // -> fcat ch 0..31
    conv1x1_wmma<<<dim3(64*64/64, NB), 256, 0, stream>>>(
        r1h, wkPyr2 + 1*32*64, pyr_b2 + 1*32, nullptr, e1f,
        64, 32, 64*64, 32*64*64, 2);
    conv1x1_wmma<<<dim3(32*32/64, NB), 256, 0, stream>>>(
        r2h, wkPyr2 + 2*32*64, pyr_b2 + 2*32, nullptr, e2f,
        64, 32, 32*32, 32*32*32, 2);

    // --- upsample evidence into fcat ch 32..63 / 64..95 ---
    bilinear_kernel<<<dim3(NPIX/256, NB*32), 256, 0, stream>>>(
        e1f, 32, 64, 64, HF, HF, fcat, 128, 32);
    bilinear_kernel<<<dim3(NPIX/256, NB*32), 256, 0, stream>>>(
        e2f, 32, 32, 32, HF, HF, fcat, 128, 64);

    // --- boundary detection ---
    channel_mean_kernel<<<NB*64, 256, 0, stream>>>(xf32, means);
    topk_kernel<<<NB, 32, 0, stream>>>(means, idxb);
    sobel_sel_kernel<<<dim3(NPIX/256, TOPK, NB), 256, 0, stream>>>(xf32, idxb, selb, mxb);
    bm_kernel<<<dim3(NPIX/256, NB), 256, 0, stream>>>(selb, mxb, scale_p, bmb, minmax);
    bm_finalize_kernel<<<(NB*NPIX)/256, 256, 0, stream>>>(bmb, minmax);

    // --- boundary-aware evidence branch (reuses r0h) ---
    conv3x3_wmma<<<dim3(2, 128, NB), 256, 0, stream>>>(
        xf16, wkBw1, bb1, r0h, HF, HF, 1, 64);
    conv1x1_wmma<<<dim3(NPIX/64, NB), 256, 0, stream>>>(
        r0h, wkBw2, bb2, bevr2, nullptr, 64, 32, NPIX, 32*NPIX, 1);
    bev3_kernel<<<dim3(NPIX/256, NB), 256, 0, stream>>>(bevr2, bw3, bb3, bmb, fcat);

    // --- fusion (K=128 padded, relu) -> ffeat in r0h ---
    conv1x1_wmma<<<dim3(NPIX/32, NB), 256, 0, stream>>>(
        fcat, wkFw1, fb1, r0h, nullptr, 128, 64, NPIX, 64*NPIX, 1);

    // --- Dirichlet head + weighted unfold ---
    final_kernel<<<dim3(NPIX/256, NB), 256, 0, stream>>>(
        r0h, fw2, fb2, uw, xf32, outProb, outTok);
}
